// DCNv2_66855460930023
// MI455X (gfx1250) — matrix operand order repaired and checked by reading, not compiled, not run on hardware
//
#include <hip/hip_runtime.h>

// ---------------------------------------------------------------------------
// DCNv2 stack for MI455X (gfx1250, wave32, WMMA).
//   - offset/mask convs + bilinear sampling: scalar fp32 (gather-bound)
//   - per-layer reduce einsum: im2col (f16) x packed weights (f16) GEMM on
//     v_wmma_f32_16x16x32_f16, f32 accumulate; N-register-blocked (NT tiles
//     per wave) so each A fetch feeds NT WMMAs (col buffer is the dominant
//     HBM intermediate -> halve/quarter its traffic).
// ---------------------------------------------------------------------------

typedef __attribute__((ext_vector_type(16))) _Float16 v16h;
typedef __attribute__((ext_vector_type(8)))  float    v8f;

// ---------------------------------------------------------------- offset/mask
// y[b,ch,ho,wo] = conv3x3_s2_p1(x, w)[ch] ; ch<18 -> raw offsets, ch>=18 ->
// 2*sigmoid(mask conv). One thread per (b,ch,ho,wo).
__global__ void dcn_offmask_conv(const float* __restrict__ x,
                                 const float* __restrict__ ow, const float* __restrict__ ob,
                                 const float* __restrict__ mw, const float* __restrict__ mb,
                                 float* __restrict__ offs, float* __restrict__ mask,
                                 int B, int C, int H, int W, int Ho, int Wo) {
  int idx = blockIdx.x * blockDim.x + threadIdx.x;
  int total = B * 27 * Ho * Wo;
  if (idx >= total) return;
  int wo = idx % Wo; int t = idx / Wo;
  int ho = t % Ho;   t /= Ho;
  int ch = t % 27;   int b = t / 27;

  bool is_mask = (ch >= 18);
  int cc = is_mask ? ch - 18 : ch;
  const float* w = (is_mask ? mw : ow) + (size_t)cc * C * 9;
  float s = is_mask ? mb[cc] : ob[cc];

  int hy = ho * 2 - 1, wx = wo * 2 - 1;
  for (int c = 0; c < C; ++c) {
    const float* xc = x + ((size_t)b * C + c) * H * W;
    const float* wc = w + c * 9;
    for (int ky = 0; ky < 3; ++ky) {
      int yy = hy + ky;
      if ((unsigned)yy >= (unsigned)H) continue;
      const float* xr = xc + (size_t)yy * W;
      for (int kx = 0; kx < 3; ++kx) {
        int xx = wx + kx;
        if ((unsigned)xx >= (unsigned)W) continue;
        s += xr[xx] * wc[ky * 3 + kx];
      }
    }
  }
  if (is_mask)
    mask[(((size_t)b * 9 + cc) * Ho + ho) * Wo + wo] = 2.0f / (1.0f + __expf(-s));
  else
    offs[(((size_t)b * 18 + cc) * Ho + ho) * Wo + wo] = s;
}

// ------------------------------------------------------------------ sampling
// For each (b,k,ho,wo): bilinear-sample all C channels at the deformed
// position, modulate by mask, store f16 into the col buffer laid out
// [m = b*Ho*Wo + ho*Wo + wo][ kidx = c*9 + k ] with row stride Kpad.
__global__ void dcn_sample_mod(const float* __restrict__ x,
                               const float* __restrict__ offs, const float* __restrict__ mask,
                               _Float16* __restrict__ col,
                               int B, int C, int H, int W, int Ho, int Wo,
                               int Kpad, int CK) {
  int idx = blockIdx.x * blockDim.x + threadIdx.x;
  int total = B * 9 * Ho * Wo;
  if (idx >= total) return;
  int wo = idx % Wo; int t = idx / Wo;
  int ho = t % Ho;   t /= Ho;
  int k  = t % 9;    int b = t / 9;

  size_t so = ((size_t)b * 18 + 2 * k) * Ho * Wo + (size_t)ho * Wo + wo;
  float dy = offs[so];
  float dx = offs[so + (size_t)Ho * Wo];          // channel 2k+1
  float mk = mask[(((size_t)b * 9 + k) * Ho + ho) * Wo + wo];

  float py = (float)(ho * 2 - 1 + (k / 3)) + dy;
  float px = (float)(wo * 2 - 1 + (k % 3)) + dx;
  float y0f = floorf(py), x0f = floorf(px);
  int yi0 = (int)y0f, xi0 = (int)x0f;
  float fy = py - y0f, fx = px - x0f;

  bool vy0 = (yi0 >= 0) && (yi0 < H), vy1 = (yi0 + 1 >= 0) && (yi0 + 1 < H);
  bool vx0 = (xi0 >= 0) && (xi0 < W), vx1 = (xi0 + 1 >= 0) && (xi0 + 1 < W);
  int ya = min(max(yi0, 0), H - 1),     yb = min(max(yi0 + 1, 0), H - 1);
  int xa = min(max(xi0, 0), W - 1),     xb = min(max(xi0 + 1, 0), W - 1);
  float w00 = (vy0 && vx0) ? (1.0f - fy) * (1.0f - fx) : 0.0f;
  float w01 = (vy0 && vx1) ? (1.0f - fy) * fx          : 0.0f;
  float w10 = (vy1 && vx0) ? fy * (1.0f - fx)          : 0.0f;
  float w11 = (vy1 && vx1) ? fy * fx                   : 0.0f;

  size_t m = (size_t)b * Ho * Wo + (size_t)ho * Wo + wo;
  _Float16* crow = col + m * Kpad;
  for (int c = 0; c < C; ++c) {
    const float* xc = x + ((size_t)b * C + c) * H * W;
    float v = xc[(size_t)ya * W + xa] * w00 + xc[(size_t)ya * W + xb] * w01 +
              xc[(size_t)yb * W + xa] * w10 + xc[(size_t)yb * W + xb] * w11;
    crow[c * 9 + k] = (_Float16)(v * mk);
  }
  if (k == 0) {                         // zero K padding (layer 1: 27 -> 32)
    for (int p = CK; p < Kpad; ++p) crow[p] = (_Float16)0.0f;
  }
}

// ------------------------------------------------------------- weight packing
// packB[tile = kt*Ntiles+nt][lane][j] = wT[k, n] in the per-lane B
// layout this code feeds v_wmma_f32_16x16x32_f16 (contiguous k, as A): n = nt*16 + (lane&15),
// k = kt*32 + (lane>=16 ? 16 : 0) + j.   wT[c*9+kk, o] = rw[o, c, kk].
__global__ void dcn_pack_weights(const float* __restrict__ rw, _Float16* __restrict__ packB,
                                 int O, int CK, int Kpad) {
  int Ntiles = O >> 4;
  int idx = blockIdx.x * blockDim.x + threadIdx.x;
  int total = (Kpad >> 5) * Ntiles * 32;
  if (idx >= total) return;
  int lane = idx & 31;
  int tile = idx >> 5;
  int nt = tile % Ntiles, kt = tile / Ntiles;
  int n  = nt * 16 + (lane & 15);
  int kb = kt * 32 + ((lane >> 4) << 4);
  _Float16* dst = packB + ((size_t)tile * 32 + lane) * 16;
  for (int j = 0; j < 16; ++j) {
    int k = kb + j;
    dst[j] = (_Float16)((k < CK) ? rw[(size_t)n * CK + k] : 0.0f);
  }
}

// ------------------------------------------------------------------ WMMA GEMM
// out[b,o,hw] = sum_k col[m,k] * wT[k,o],  m = b*HoWo + hw.
// One wave per 16x(16*NT) output block: A (16xKpad rows) loaded once per
// k-step, reused for NT accumulators -> NT v_wmma per 2 b128 A-loads.
template <int NT>
__global__ __launch_bounds__(32)
void dcn_gemm_wmma(const _Float16* __restrict__ col, const _Float16* __restrict__ packB,
                   float* __restrict__ out, int M, int Kpad, int O, int HoWo) {
  int mt  = blockIdx.x;
  int ntg = blockIdx.y * NT;      // first of NT consecutive n-tiles
  int lane  = threadIdx.x;
  int khalf = lane >> 4;          // 0: lanes 0-15, 1: lanes 16-31
  int rsel  = lane & 15;
  int Ntiles = O >> 4;

  int row = mt * 16 + rsel;       // A row held by this lane
  int rowc = (row < M) ? row : 0; // clamp (dead tile rows are masked on store)
  const _Float16* arow = col + (size_t)rowc * Kpad;
  // per-(k-tile,n-tile) block of 32 lanes x 16 f16 = 512 f16
  const _Float16* bbase = packB + ((size_t)ntg * 32 + lane) * 16;
  size_t bkstride = (size_t)Ntiles * 512;   // k-tile stride in packB

  v8f acc[NT];
#pragma unroll
  for (int q = 0; q < NT; ++q) acc[q] = (v8f){};

  int Ksteps = Kpad >> 5;
  for (int kt = 0; kt < Ksteps; ++kt) {
    int k0 = kt * 32 + khalf * 16;
    v16h a;
    // A per-lane layout: elems 0..15 <- K k0..k0+15
    ((uint4*)&a)[0] = *(const uint4*)(arow + k0);
    ((uint4*)&a)[1] = *(const uint4*)(arow + k0 + 8);
    __builtin_prefetch(arow + k0 + 32, 0, 1);   // global_prefetch next k-step
    const _Float16* bk = bbase + (size_t)kt * bkstride;
#pragma unroll
    for (int q = 0; q < NT; ++q) {
      v16h bm = *(const v16h*)(bk + q * 512);
      acc[q] = __builtin_amdgcn_wmma_f32_16x16x32_f16(
          /*neg_a=*/false, a, /*neg_b=*/false, bm,
          /*c_mod=*/(short)0, acc[q], /*reuse_a=*/false, /*reuse_b=*/false);
    }
  }

  // C/D scatter: lane holds (M = v + 8*khalf, N = rsel) of each tile.
#pragma unroll
  for (int q = 0; q < NT; ++q) {
    int ncol = (ntg + q) * 16 + rsel;
    for (int v = 0; v < 8; ++v) {
      int m = mt * 16 + v + 8 * khalf;
      if (m < M) {
        int b = m / HoWo, hw = m - b * HoWo;
        out[((size_t)b * O + ncol) * HoWo + hw] = acc[q][v];
      }
    }
  }
}

// ----------------------------------------------------------------- pool/heads
__global__ void dcn_pool(const float* __restrict__ x, float* __restrict__ pool,
                         int B, int C, int HoWo) {
  int idx = blockIdx.x * blockDim.x + threadIdx.x;
  if (idx >= B * C) return;
  const float* p = x + (size_t)idx * HoWo;
  float s = 0.0f;
  for (int i = 0; i < HoWo; ++i) s += p[i];
  pool[idx] = s / (float)HoWo;
}

__global__ void dcn_heads(const float* __restrict__ pool,
                          const float* __restrict__ wo, const float* __restrict__ bo,
                          const float* __restrict__ wsh, const float* __restrict__ bs,
                          const float* __restrict__ we, const float* __restrict__ be,
                          float* __restrict__ out, int B) {
  int idx = blockIdx.x * blockDim.x + threadIdx.x;
  if (idx >= B * 62) return;
  int n = idx % 62, b = idx / 62;
  const float* w; float bias;
  if (n < 12)      { w = wo  + (size_t)n * 1280;        bias = bo[n]; }
  else if (n < 52) { w = wsh + (size_t)(n - 12) * 1280; bias = bs[n - 12]; }
  else             { w = we  + (size_t)(n - 52) * 1280; bias = be[n - 52]; }
  const float* p = pool + (size_t)b * 1280;
  float s = bias;
  for (int c = 0; c < 1280; ++c) s += p[c] * w[c];
  out[(size_t)b * 62 + n] = s;
}

// ---------------------------------------------------------------------- host
extern "C" void kernel_launch(void* const* d_in, const int* in_sizes, int n_in,
                              void* d_out, int out_size, void* d_ws, size_t ws_size,
                              hipStream_t stream) {
  (void)in_sizes; (void)n_in; (void)out_size; (void)ws_size;
  const float* x0 = (const float*)d_in[0];
  const int cins[7]  = {3, 32, 32, 32, 32, 32, 32};
  const int couts[7] = {32, 32, 32, 32, 32, 32, 1280};
  const int B = 64;

  // Workspace arena (peak ~270 MB):
  //   XA (100 MB) | XB (32 MB) | COL+OFFMASK arena (136 MB) | packed weights
  char* ws = (char*)d_ws;
  float* XA    = (float*)(ws);
  float* XB    = (float*)(ws + ((size_t)100 << 20));
  char*  COLOM = ws + ((size_t)132 << 20);
  _Float16* WPK = (_Float16*)(ws + ((size_t)268 << 20));

  // Pack all reduce weights into WMMA B-tile layout (independent of layers).
  size_t wofsArr[7]; size_t wofs = 0;
  for (int l = 0; l < 7; ++l) {
    int CK = cins[l] * 9;
    int Kpad = (CK + 31) & ~31;
    int O = couts[l];
    wofsArr[l] = wofs;
    const float* rw = (const float*)d_in[1 + l * 5 + 4];
    int total = (Kpad >> 5) * (O >> 4) * 32;
    dcn_pack_weights<<<(total + 255) / 256, 256, 0, stream>>>(rw, WPK + wofs, O, CK, Kpad);
    wofs += (size_t)Kpad * O;
  }

  const float* xin = x0;
  int H = 224, W = 224;
  for (int l = 0; l < 7; ++l) {
    int C = cins[l], O = couts[l];
    int Ho = (H - 1) / 2 + 1, Wo = (W - 1) / 2 + 1;
    int CK = C * 9, Kpad = (CK + 31) & ~31;
    int HoWo = Ho * Wo, M = B * HoWo;
    float* xout = (l & 1) ? XB : XA;

    _Float16* col = (_Float16*)COLOM;
    size_t colBytes = (((size_t)M * Kpad * 2) + 255) & ~(size_t)255;
    float* offs = (float*)(COLOM + colBytes);
    float* mask = offs + (size_t)B * 18 * HoWo;

    const float* ow = (const float*)d_in[1 + l * 5 + 0];
    const float* ob = (const float*)d_in[1 + l * 5 + 1];
    const float* mw = (const float*)d_in[1 + l * 5 + 2];
    const float* mb = (const float*)d_in[1 + l * 5 + 3];

    int t1 = B * 27 * HoWo;
    dcn_offmask_conv<<<(t1 + 255) / 256, 256, 0, stream>>>(
        xin, ow, ob, mw, mb, offs, mask, B, C, H, W, Ho, Wo);

    int t2 = B * 9 * HoWo;
    dcn_sample_mod<<<(t2 + 255) / 256, 256, 0, stream>>>(
        xin, offs, mask, col, B, C, H, W, Ho, Wo, Kpad, CK);

    int Ntiles = O >> 4;
    unsigned mtiles = (unsigned)((M + 15) / 16);
    if ((Ntiles & 3) == 0) {        // layer 7: O=1280 -> 20 groups of 4
      dim3 gg(mtiles, (unsigned)(Ntiles / 4));
      dcn_gemm_wmma<4><<<gg, 32, 0, stream>>>(col, WPK + wofsArr[l], xout, M, Kpad, O, HoWo);
    } else {                        // layers 1-6: O=32 -> 1 group of 2
      dim3 gg(mtiles, (unsigned)(Ntiles / 2));
      dcn_gemm_wmma<2><<<gg, 32, 0, stream>>>(col, WPK + wofsArr[l], xout, M, Kpad, O, HoWo);
    }

    xin = xout; H = Ho; W = Wo;
  }

  // Outputs: [0, 64*62) = heads concat, [64*62, 64*62 + 64*1280) = pool_x.
  float* outp = (float*)d_out;
  float* pool = outp + 64 * 62;
  int HoWo = H * W;   // 2*2 = 4
  int tp = B * 1280;
  dcn_pool<<<(tp + 255) / 256, 256, 0, stream>>>(xin, pool, B, 1280, HoWo);
  int th = B * 62;
  dcn_heads<<<(th + 255) / 256, 256, 0, stream>>>(
      pool, (const float*)d_in[36], (const float*)d_in[37],
      (const float*)d_in[38], (const float*)d_in[39],
      (const float*)d_in[40], (const float*)d_in[41], outp, B);
}
